// TensorConvLayer_20607253086905
// MI455X (gfx1250) — matrix operand order repaired and checked by reading, not compiled, not run on hardware
//
#include <hip/hip_runtime.h>
#include <math.h>

#define N_NODES 100000
#define N_EDGES 500000
#define MUL0 16
#define MUL1 8
#define FEAT 40          // MUL0 + 3*MUL1
#define H 64             // H_EDGE
#define WN 576           // W_NUMEL
#define TILE_E 64
#define EPS_BN 1e-5f

typedef __attribute__((ext_vector_type(16))) _Float16     v16h;
typedef __attribute__((ext_vector_type(8)))  _Float16     v8h;
typedef __attribute__((ext_vector_type(8)))  float        v8f;
typedef __attribute__((ext_vector_type(4)))  unsigned int v4u;
typedef __attribute__((ext_vector_type(8)))  int          v8i;
typedef __attribute__((ext_vector_type(4)))  int          v4i;

#if defined(__gfx1250__) && __has_builtin(__builtin_amdgcn_tensor_load_to_lds) && \
    __has_builtin(__builtin_amdgcn_s_wait_tensorcnt)
#define HAVE_TDM 1
#else
#define HAVE_TDM 0
#endif

// ---- vector load helpers ----------------------------------------------------
__device__ inline v16h ld16h(const _Float16* p) {     // 16 contiguous halfs
  union { v16h v; float4 f[2]; } u;
  u.f[0] = *(const float4*)(p);
  u.f[1] = *(const float4*)(p + 8);
  return u.v;
}
__device__ inline v8h ld8h(const _Float16* p) {       // 8 contiguous halfs
  union { v8h v; float4 f; } u;
  u.f = *(const float4*)(p);
  return u.v;
}

// ---- WMMA fragment gathers (layouts per CDNA5 ISA 7.12.2, wave32) ----------

// A fragment: 16 (rows m) x 32 (k), f32 row-major LDS tile.  Lane pattern is
// two contiguous 8-element runs: k in [ko,ko+8) and [ko+16,ko+24).
__device__ inline v16h frag_a_lds_f32(const float* base, int lda, int k0, int lane) {
  const int m  = lane & 15;
  const int ko = k0 + ((lane & 16) ? 8 : 0);
  const float* p = base + m * lda + ko;
  const float4 f0 = *(const float4*)(p);
  const float4 f1 = *(const float4*)(p + 4);
  const float4 f2 = *(const float4*)(p + 16);
  const float4 f3 = *(const float4*)(p + 20);
  v16h a;
  a[0]  = (_Float16)f0.x; a[1]  = (_Float16)f0.y;
  a[2]  = (_Float16)f0.z; a[3]  = (_Float16)f0.w;
  a[4]  = (_Float16)f1.x; a[5]  = (_Float16)f1.y;
  a[6]  = (_Float16)f1.z; a[7]  = (_Float16)f1.w;
  a[8]  = (_Float16)f2.x; a[9]  = (_Float16)f2.y;
  a[10] = (_Float16)f2.z; a[11] = (_Float16)f2.w;
  a[12] = (_Float16)f3.x; a[13] = (_Float16)f3.y;
  a[14] = (_Float16)f3.z; a[15] = (_Float16)f3.w;
  return a;
}

// A fragment from an f16 row-major LDS tile: 2x ds_load_b128.
__device__ inline v16h frag_a_lds_f16(const _Float16* base, int lda, int k0, int lane) {
  const int m  = lane & 15;
  const int ko = k0 + ((lane & 16) ? 8 : 0);
  const _Float16* p = base + m * lda + ko;
  union { v16h v; float4 f[2]; } u;
  u.f[0] = *(const float4*)(p);
  u.f[1] = *(const float4*)(p + 16);
  return u.v;
}

// ---- K0: zero workspace -----------------------------------------------------
__global__ void zero_f32(float* __restrict__ p, long long n) {
  long long i = (long long)blockIdx.x * blockDim.x + threadIdx.x;
  const long long stride = (long long)gridDim.x * blockDim.x;
  for (; i < n; i += stride) p[i] = 0.0f;
}

// ---- K0b: pack f32 weight matrix into fragment-major f16 B layout ----------
// packed[((tile*2 + kstep)*32 + lane)*16 + i] =
//   w[(kstep*32 + (lane&16?8:0) + (i<8?i:i+8)) * ldb + tile*16 + (lane&15)]
__global__ void pack_b(const float* __restrict__ w, _Float16* __restrict__ pk,
                       int ldb, int total) {
  const int idx = blockIdx.x * blockDim.x + threadIdx.x;
  if (idx >= total) return;
  const int i    = idx & 15;
  const int lane = (idx >> 4) & 31;
  const int s    = (idx >> 9) & 1;
  const int t    = idx >> 10;
  const int k = s * 32 + ((lane & 16) ? 8 : 0) + (i < 8 ? i : i + 8);
  const int n = t * 16 + (lane & 15);
  pk[idx] = (_Float16)w[k * ldb + n];
}

// ---- K1: fused edge MLP (WMMA) + tensor product + scatter -------------------
__global__ __launch_bounds__(128) void edge_mlp_tp_scatter(
    const float* __restrict__ atom, const float* __restrict__ ef,
    const float* __restrict__ esh,  const int*   __restrict__ eidx,
    const v16h* __restrict__ w1pk,  const float* __restrict__ b1,
    const v16h* __restrict__ w2pk,  const float* __restrict__ b2,
    float* __restrict__ sums,       float* __restrict__ counts) {
  __shared__ float    sXf[TILE_E * H];   // edge features, f32 (TDM dest, 16 KB)
  __shared__ _Float16 sY[TILE_E * H];    // hidden relu acts, f16       (8 KB)
  __shared__ _Float16 sW[TILE_E * WN];   // per-edge weights, f16      (72 KB)

  const int tid  = threadIdx.x;
  const int lane = tid & 31;
  const int wid  = tid >> 5;                   // 4 waves
  const int e0   = blockIdx.x * TILE_E;

#if HAVE_TDM
  // Tensor Data Mover: DMA the 64x64 f32 edge tile into LDS. Rows past the
  // end of the edge array zero-fill via the D# OOB rule (tensor_dim1 =
  // remaining rows). Wave 0 issues + waits (TENSORcnt is per-wave state);
  // the block barrier releases the other waves.
  if (tid < 32) {
    const unsigned long long gaddr =
        (unsigned long long)(uintptr_t)(ef + (size_t)e0 * H);
    const unsigned int ldsaddr = (unsigned int)(uintptr_t)(&sXf[0]);
    const unsigned int tdim1   = (unsigned int)(N_EDGES - e0);   // rows avail

    v4u g0;
    g0[0] = 1u;                                   // count=1 (valid user D#)
    g0[1] = ldsaddr;                              // lds_addr [63:32]
    g0[2] = (unsigned int)(gaddr & 0xFFFFFFFFu);  // global_addr [95:64]
    g0[3] = (unsigned int)((gaddr >> 32) & 0x1FFFFFFu) | (2u << 30); // type=2

    v8i g1;
    g1[0] = (int)(2u << 16);                      // data_size = 4 bytes
    g1[1] = (int)((H & 0xFFFFu) << 16);           // tensor_dim0[15:0]
    g1[2] = (int)(((H >> 16) & 0xFFFFu) | ((tdim1 & 0xFFFFu) << 16));
    g1[3] = (int)(((tdim1 >> 16) & 0xFFFFu) | ((unsigned)TILE_E << 16)); // tile_dim0
    g1[4] = (int)TILE_E;                          // tile_dim1=64, tile_dim2=0
    g1[5] = (int)H;                               // tensor_dim0_stride lo32
    g1[6] = 0;
    g1[7] = 0;

    v4i gz = {0, 0, 0, 0};
#if __clang_major__ >= 23
    v8i gz8 = {0, 0, 0, 0, 0, 0, 0, 0};
    __builtin_amdgcn_tensor_load_to_lds(g0, g1, gz, gz, gz8, 0);
#else
    __builtin_amdgcn_tensor_load_to_lds(g0, g1, gz, gz, 0);
#endif
    __builtin_amdgcn_s_wait_tensorcnt(0);
  }
#else
  // Fallback: coalesced float4 staging.
  {
    const int r = tid >> 1;
    const int h = (tid & 1) * 32;
    const int e = e0 + r;
    if (e < N_EDGES) {
      const float4* src = (const float4*)(ef + (size_t)e * H + h);
#pragma unroll
      for (int q = 0; q < 8; ++q)
        *(float4*)(sXf + r * H + h + q * 4) = src[q];
    } else {
#pragma unroll
      for (int q = 0; q < 32; ++q) sXf[r * H + h + q] = 0.0f;
    }
  }
#endif
  __syncthreads();

  const int rowbase = wid * 16;                // this wave's 16-edge strip
  const int hi8 = (lane & 16) ? 8 : 0;
  const int nn  = lane & 15;

  // GEMM1: Y = relu(X @ W1 + b1)   (16x64 strip, 8 WMMAs)
  {
    v16h a0 = frag_a_lds_f32(sXf + rowbase * H, H, 0,  lane);
    v16h a1 = frag_a_lds_f32(sXf + rowbase * H, H, 32, lane);
#pragma unroll
    for (int t = 0; t < 4; ++t) {
      const int n0 = t * 16;
      v8f acc = {};
      v16h bA = w1pk[(t * 2 + 0) * 32 + lane];
      v16h bB = w1pk[(t * 2 + 1) * 32 + lane];
      acc = __builtin_amdgcn_wmma_f32_16x16x32_f16(false, a0, false, bA,
                                                   (short)0, acc, false, false);
      acc = __builtin_amdgcn_wmma_f32_16x16x32_f16(false, a1, false, bB,
                                                   (short)0, acc, false, false);
      const float bias = b1[n0 + nn];
#pragma unroll
      for (int r = 0; r < 8; ++r) {
        float y = acc[r] + bias;
        y = y > 0.0f ? y : 0.0f;
        sY[(rowbase + r + hi8) * H + n0 + nn] = (_Float16)y;
      }
    }
  }
  __syncthreads();

  // GEMM2: W = Y @ W2 + b2         (16x576 strip, 72 WMMAs, A reused)
  {
    v16h a0 = frag_a_lds_f16(sY + rowbase * H, H, 0,  lane);
    v16h a1 = frag_a_lds_f16(sY + rowbase * H, H, 32, lane);
    for (int t = 0; t < 36; ++t) {
      const int n0 = t * 16;
      v8f acc = {};
      v16h bA = w2pk[(t * 2 + 0) * 32 + lane];
      v16h bB = w2pk[(t * 2 + 1) * 32 + lane];
      acc = __builtin_amdgcn_wmma_f32_16x16x32_f16(false, a0, false, bA,
                                                   (short)0, acc, false, false);
      acc = __builtin_amdgcn_wmma_f32_16x16x32_f16(false, a1, false, bB,
                                                   (short)0, acc, false, false);
      const float bias = b2[n0 + nn];
#pragma unroll
      for (int r = 0; r < 8; ++r)
        sW[(rowbase + r + hi8) * WN + n0 + nn] = (_Float16)(acc[r] + bias);
    }
  }
  __syncthreads();

  // Tensor product + scatter: 2 threads per edge, vectorized row loads.
  const int el   = tid >> 1;
  const int part = tid & 1;
  const int e    = e0 + el;
  if (e >= N_EDGES) return;

  const int dst = eidx[e];                 // edge_index[0]
  const int src = eidx[N_EDGES + e];       // edge_index[1]
  const float* x = atom + (size_t)dst * FEAT;
  const float sh0 = esh[(size_t)e * 4 + 0];
  const float s1x = esh[(size_t)e * 4 + 1];
  const float s1y = esh[(size_t)e * 4 + 2];
  const float s1z = esh[(size_t)e * 4 + 3];
  const _Float16* we = sW + el * WN;       // 1152 B stride: 16B aligned

  const float inv_s2 = 0.70710678118654752f;
  const float c00 = inv_s2 * 0.25f;                 // 1/sqrt(MUL0)
  const float c10 = inv_s2 * 0.20412414523193154f;  // 1/sqrt(MUL1*3)
  const float c11 = inv_s2 * 0.35355339059327373f;  // 1/sqrt(MUL1)
  const float c01 = inv_s2 * 0.25f;                 // 1/sqrt(MUL0)

  float x0[MUL0];
#pragma unroll
  for (int i = 0; i < MUL0; ++i) x0[i] = x[i];

  if (part == 0) {
    // out0_j = sh0*c00 * (x0^T W00)_j + c10 * ((x1.sh1)^T W10)_j
    float y1[MUL1];
#pragma unroll
    for (int i = 0; i < MUL1; ++i)
      y1[i] = x[MUL0 + i * 3 + 0] * s1x + x[MUL0 + i * 3 + 1] * s1y +
              x[MUL0 + i * 3 + 2] * s1z;
    float acc[MUL0];
#pragma unroll
    for (int j = 0; j < MUL0; ++j) acc[j] = 0.0f;
    const float sA = sh0 * c00;
#pragma unroll
    for (int i = 0; i < MUL0; ++i) {
      const v16h row = ld16h(we + i * 16);       // W00 row i
      const float f = x0[i] * sA;
#pragma unroll
      for (int j = 0; j < MUL0; ++j) acc[j] += (float)row[j] * f;
    }
#pragma unroll
    for (int i = 0; i < MUL1; ++i) {
      const v16h row = ld16h(we + 448 + i * 16); // W10 row i
      const float f = y1[i] * c10;
#pragma unroll
      for (int j = 0; j < MUL0; ++j) acc[j] += (float)row[j] * f;
    }
#pragma unroll
    for (int j = 0; j < MUL0; ++j)
      atomicAdd(&sums[(size_t)src * FEAT + j], acc[j]);
    atomicAdd(&counts[src], 1.0f);
  } else {
    // out1_{j,c} = sh0*c11 * (x1^T W11)_{j,c} + c01*sh1_c * (x0^T W01)_j
    float x1[MUL1][3];
#pragma unroll
    for (int i = 0; i < MUL1; ++i)
#pragma unroll
      for (int c = 0; c < 3; ++c) x1[i][c] = x[MUL0 + i * 3 + c];
    float d[MUL1];
    float m[MUL1][3];
#pragma unroll
    for (int j = 0; j < MUL1; ++j) {
      d[j] = 0.0f; m[j][0] = 0.0f; m[j][1] = 0.0f; m[j][2] = 0.0f;
    }
#pragma unroll
    for (int i = 0; i < MUL0; ++i) {
      const v8h row = ld8h(we + 320 + i * 8);    // W01 row i
      const float f = x0[i];
#pragma unroll
      for (int j = 0; j < MUL1; ++j) d[j] += (float)row[j] * f;
    }
#pragma unroll
    for (int i = 0; i < MUL1; ++i) {
      const v8h row = ld8h(we + 256 + i * 8);    // W11 row i
#pragma unroll
      for (int j = 0; j < MUL1; ++j) {
        const float wv = (float)row[j];
        m[j][0] += wv * x1[i][0];
        m[j][1] += wv * x1[i][1];
        m[j][2] += wv * x1[i][2];
      }
    }
    const float ms = sh0 * c11;
#pragma unroll
    for (int j = 0; j < MUL1; ++j) {
      const float dd = d[j] * c01;
      const size_t o = (size_t)src * FEAT + MUL0 + j * 3;
      atomicAdd(&sums[o + 0], m[j][0] * ms + dd * s1x);
      atomicAdd(&sums[o + 1], m[j][1] * ms + dd * s1y);
      atomicAdd(&sums[o + 2], m[j][2] * ms + dd * s1z);
    }
  }
}

// ---- K2: segment mean + residual (in-place) + global stat accumulation -----
__global__ __launch_bounds__(256) void node_combine(
    const float* __restrict__ atom, float* __restrict__ sums,
    const float* __restrict__ counts, float* __restrict__ stats) {
  __shared__ float st[40];  // [0,16): sum_s  [16,32): sumsq_s  [32,40): sumsq_v
  if (threadIdx.x < 40) st[threadIdx.x] = 0.0f;
  __syncthreads();

  const int n = blockIdx.x * blockDim.x + threadIdx.x;
  if (n < N_NODES) {
    float cnt = counts[n];
    cnt = cnt > 1.0f ? cnt : 1.0f;
    const float inv = 1.0f / cnt;
    float pre[FEAT];
#pragma unroll
    for (int f = 0; f < FEAT; ++f) {
      const size_t idx = (size_t)n * FEAT + f;
      const float v = sums[idx] * inv + atom[idx];
      pre[f] = v;
      sums[idx] = v;  // stash pre-norm activations
    }
#pragma unroll
    for (int j = 0; j < MUL0; ++j) {
      atomicAdd(&st[j], pre[j]);
      atomicAdd(&st[16 + j], pre[j] * pre[j]);
    }
#pragma unroll
    for (int i = 0; i < MUL1; ++i) {
      const float a = pre[MUL0 + i * 3 + 0];
      const float b = pre[MUL0 + i * 3 + 1];
      const float c = pre[MUL0 + i * 3 + 2];
      atomicAdd(&st[32 + i], a * a + b * b + c * c);
    }
  }
  __syncthreads();
  if (threadIdx.x < 40) atomicAdd(&stats[threadIdx.x], st[threadIdx.x]);
}

// ---- K3: apply batch-norm style normalization -------------------------------
__global__ __launch_bounds__(256) void node_norm(
    const float* __restrict__ sums, const float* __restrict__ stats,
    const float* __restrict__ bnw,  const float* __restrict__ bnb,
    float* __restrict__ out) {
  const int n = blockIdx.x * blockDim.x + threadIdx.x;
  if (n >= N_NODES) return;
  const float invN = 1.0f / (float)N_NODES;
#pragma unroll
  for (int j = 0; j < MUL0; ++j) {
    const float mean = stats[j] * invN;
    const float ex2  = stats[16 + j] * invN;
    const float var  = ex2 - mean * mean;
    const float sc   = bnw[j] * rsqrtf(var + EPS_BN);
    const size_t idx = (size_t)n * FEAT + j;
    out[idx] = (sums[idx] - mean) * sc + bnb[j];
  }
#pragma unroll
  for (int i = 0; i < MUL1; ++i) {
    const float vn = stats[32 + i] * (invN * (1.0f / 3.0f));
    const float sc = bnw[MUL0 + i] * rsqrtf(vn + EPS_BN);
#pragma unroll
    for (int c = 0; c < 3; ++c) {
      const size_t idx = (size_t)n * FEAT + MUL0 + i * 3 + c;
      out[idx] = sums[idx] * sc;
    }
  }
}

// ---- host launch ------------------------------------------------------------
extern "C" void kernel_launch(void* const* d_in, const int* in_sizes, int n_in,
                              void* d_out, int out_size, void* d_ws,
                              size_t ws_size, hipStream_t stream) {
  const float* atom = (const float*)d_in[0];
  const float* ef   = (const float*)d_in[1];
  const float* esh  = (const float*)d_in[2];
  const int*   eidx = (const int*)d_in[3];
  const float* w1   = (const float*)d_in[4];
  const float* b1   = (const float*)d_in[5];
  const float* w2   = (const float*)d_in[6];
  const float* b2   = (const float*)d_in[7];
  const float* bnw  = (const float*)d_in[8];
  const float* bnb  = (const float*)d_in[9];
  float* out = (float*)d_out;

  // workspace: sums[N*40] | counts[N] | stats[40] | w1pk[4096h] | w2pk[36864h]
  float* sums   = (float*)d_ws;
  float* counts = sums + (size_t)N_NODES * FEAT;
  float* stats  = counts + N_NODES;
  _Float16* w1pk = (_Float16*)(stats + 40);     // 32B-aligned offset
  _Float16* w2pk = w1pk + (size_t)H * H;
  const long long ztotal = (long long)N_NODES * FEAT + N_NODES + 40;

  zero_f32<<<dim3(1024), dim3(256), 0, stream>>>(sums, ztotal);
  pack_b<<<dim3((H * H + 255) / 256), dim3(256), 0, stream>>>(w1, w1pk, H,
                                                              H * H);
  pack_b<<<dim3((H * WN + 255) / 256), dim3(256), 0, stream>>>(w2, w2pk, WN,
                                                               H * WN);

  const int nblk_e = (N_EDGES + TILE_E - 1) / TILE_E;
  edge_mlp_tp_scatter<<<dim3(nblk_e), dim3(128), 0, stream>>>(
      atom, ef, esh, eidx, (const v16h*)w1pk, b1, (const v16h*)w2pk, b2, sums,
      counts);

  const int nblk_n = (N_NODES + 255) / 256;
  node_combine<<<dim3(nblk_n), dim3(256), 0, stream>>>(atom, sums, counts, stats);
  node_norm<<<dim3(nblk_n), dim3(256), 0, stream>>>(sums, stats, bnw, bnb, out);
}
